// AlexNet_75600014344540
// MI455X (gfx1250) — matrix operand order repaired and checked by reading, not compiled, not run on hardware
//
#include <hip/hip_runtime.h>

typedef __attribute__((ext_vector_type(16))) _Float16 v16h;
typedef __attribute__((ext_vector_type(8)))  _Float16 v8h;
typedef __attribute__((ext_vector_type(8)))  float    v8f;

// Q1.7 quant replicating the reference exactly, including the (s|8)&248 quirk
// on negative inputs (int32 semantics identical to jnp).
__device__ __forceinline__ float quantf(float x) {
  float f = floorf(x * 128.0f);
  f = fminf(fmaxf(f, -128.0f), 127.0f);
  int s = (int)f;
  s = (s | 8) & 248;
  return (float)s * 0.0078125f;
}

__device__ __forceinline__ v8f wmma16(v16h a, v16h b, v8f c) {
  return __builtin_amdgcn_wmma_f32_16x16x32_f16(false, a, false, b, (short)0, c,
                                                false, false);
}

// ---------------------------------------------------------------------------
// Weight repack: OIHW fp32 -> A-fragment f16 layout [g][ky][kx][cc][lane][half]
// A frag (16x32 f16): lane L holds row M = L&15; half h -> K =
//   ((L>>4)<<4) + h   (contiguous, matches B fragment loads)
// ---------------------------------------------------------------------------
__global__ void repack_conv_w(const float* __restrict__ w, _Float16* __restrict__ out,
                              int Cin, int Cout) {
  long idx = (long)blockIdx.x * 256 + threadIdx.x;
  long total = (long)Cout * 9 * Cin;
  if (idx >= total) return;
  int h = (int)(idx & 15);
  int l = (int)((idx >> 4) & 31);
  long rest = idx >> 9;
  int cch = Cin >> 5;
  int cc = (int)(rest % cch); rest /= cch;
  int kx = (int)(rest % 3);   rest /= 3;
  int ky = (int)(rest % 3);
  int g  = (int)(rest / 3);
  int cout = g * 16 + (l & 15);
  int kk = ((l >> 4) << 4) + h;
  int cin = cc * 32 + kk;
  out[idx] = (_Float16)w[(((long)cout * Cin + cin) * 3 + ky) * 3 + kx];
}

// FC weights [Cout,K] fp32 -> A-fragment layout [g][cc][lane][half], zero-pad cout
__global__ void repack_fc_w(const float* __restrict__ w, _Float16* __restrict__ out,
                            int K, int Cout, int CoutP) {
  long idx = (long)blockIdx.x * 256 + threadIdx.x;
  long total = (long)CoutP * K;
  if (idx >= total) return;
  int h = (int)(idx & 15);
  int l = (int)((idx >> 4) & 31);
  long rest = idx >> 9;
  int kch = K >> 5;
  int cc = (int)(rest % kch);
  int g  = (int)(rest / kch);
  int cout = g * 16 + (l & 15);
  int kk = ((l >> 4) << 4) + h;
  int k = cc * 32 + kk;
  out[idx] = (cout < Cout) ? (_Float16)w[(long)cout * K + k] : (_Float16)0.0f;
}

// ---------------------------------------------------------------------------
// Zero the 1-pixel border frame of a padded NHWC buffer [B, H+2, W+2, C].
// Producers write only the interior; this keeps SAME-conv reads branch-free.
// ---------------------------------------------------------------------------
__global__ void zero_border(_Float16* __restrict__ buf, int H, int W, int C, long total) {
  long idx = (long)blockIdx.x * 256 + threadIdx.x;
  if (idx >= total) return;
  int Wp = W + 2, Hp = H + 2;
  int c = (int)(idx % C); long t = idx / C;
  int per = 2 * Wp + 2 * H;              // border pixels per image
  int p = (int)(t % per); int b = (int)(t / per);
  int y, xx;
  if (p < Wp)            { y = 0;      xx = p; }
  else if (p < 2 * Wp)   { y = Hp - 1; xx = p - Wp; }
  else { int q = p - 2 * Wp; y = 1 + (q >> 1); xx = (q & 1) ? (Wp - 1) : 0; }
  buf[(((long)b * Hp + y) * Wp + xx) * C + c] = (_Float16)0.0f;
}

// ---------------------------------------------------------------------------
// conv1: Cin=1, K=9 -> direct scalar conv (1.2 GFLOP, negligible).
// x: [256,64,64] fp32 ; out: padded NHWC [256,66,66,64] f16 (interior)
// ---------------------------------------------------------------------------
__global__ void conv1_direct(const float* __restrict__ x, const float* __restrict__ w,
                             const float* __restrict__ bias, _Float16* __restrict__ out) {
  long idx = (long)blockIdx.x * 256 + threadIdx.x;   // 256*64*64*64
  int co = (int)(idx & 63); long t = idx >> 6;
  int xx = (int)(t & 63); t >>= 6;
  int y  = (int)(t & 63);
  int b  = (int)(t >> 6);
  float s = bias[co];
  #pragma unroll
  for (int ky = 0; ky < 3; ++ky) {
    int iy = y + ky - 1;
    if (iy < 0 || iy >= 64) continue;
    #pragma unroll
    for (int kx = 0; kx < 3; ++kx) {
      int ix = xx + kx - 1;
      if (ix < 0 || ix >= 64) continue;
      s += x[((long)b * 64 + iy) * 64 + ix] * w[co * 9 + ky * 3 + kx];
    }
  }
  s = fmaxf(s, 0.0f);
  out[(((long)b * 66 + y + 1) * 66 + xx + 1) * 64 + co] = (_Float16)quantf(s);
}

// ---------------------------------------------------------------------------
// Implicit-GEMM 3x3 SAME conv via WMMA, 2x2 register tile per wave:
// 2 Cout groups x 2 pixel subtiles -> 4 WMMAs per (2 A + 2 B) fragment loads.
// in : padded NHWC [B,H+2,W+2,Cin] ; out: padded NHWC [B,H+2,W+2,Cout]
// grid: (N*H*W/32, Cout/32), block: 32 (one wave).
// ---------------------------------------------------------------------------
__global__ __launch_bounds__(32) void conv3x3_wmma(
    const _Float16* __restrict__ in, const _Float16* __restrict__ wpk,
    const float* __restrict__ bias, _Float16* __restrict__ out,
    int H, int W, int Cin, int Cout) {
  const int lane = threadIdx.x;
  const int g0 = blockIdx.y * 2;
  const int n = lane & 15;
  const int koff = (lane >> 4) << 4;     // B frag: lanes 0-15 K 0..15, lanes 16-31 K 16..31
  const int cch = Cin >> 5;
  const int HW = H * W;
  const int Wp = W + 2, Hp = H + 2;

  long p0 = (long)blockIdx.x * 32;       // 32 pixels = two 16-px subtiles
  long p1 = p0 + 16;
  int b0 = (int)(p0 / HW); int r0 = (int)(p0 - (long)b0 * HW);
  int y0 = r0 / W;         int x0 = r0 - y0 * W + n;
  int b1 = (int)(p1 / HW); int r1 = (int)(p1 - (long)b1 * HW);
  int y1 = r1 / W;         int x1 = r1 - y1 * W + n;

  long base0 = (((long)b0 * Hp + y0) * Wp + x0) * Cin + koff;
  long base1 = (((long)b1 * Hp + y1) * Wp + x1) * Cin + koff;

  v8f acc00 = {}, acc01 = {}, acc10 = {}, acc11 = {};
  const long gstride = (long)9 * cch * 512;
  const _Float16* aptr0 = wpk + (long)g0 * gstride + lane * 16;
  const _Float16* aptr1 = aptr0 + gstride;
  for (int ky = 0; ky < 3; ++ky) {
    for (int kx = 0; kx < 3; ++kx) {
      long o0 = base0 + ((long)ky * Wp + kx) * Cin;
      long o1 = base1 + ((long)ky * Wp + kx) * Cin;
      for (int cc = 0; cc < cch; ++cc) {
        v16h a0  = *(const v16h*)aptr0;  aptr0 += 512;
        v16h a1  = *(const v16h*)aptr1;  aptr1 += 512;
        v16h bf0 = *(const v16h*)(in + o0 + ((long)cc << 5));
        v16h bf1 = *(const v16h*)(in + o1 + ((long)cc << 5));
        acc00 = wmma16(a0, bf0, acc00);
        acc01 = wmma16(a0, bf1, acc01);
        acc10 = wmma16(a1, bf0, acc10);
        acc11 = wmma16(a1, bf1, acc11);
      }
    }
  }
  // Epilogue: bias + relu + quant; each lane writes 8 contiguous channels (16B).
  const int moff = (lane >> 4) << 3;
  long po0 = (((long)b0 * Hp + y0 + 1) * Wp + x0 + 1) * Cout;
  long po1 = (((long)b1 * Hp + y1 + 1) * Wp + x1 + 1) * Cout;
  #pragma unroll
  for (int gg = 0; gg < 2; ++gg) {
    const int cb = (g0 + gg) * 16 + moff;
    const v8f& a0r = gg ? acc10 : acc00;
    const v8f& a1r = gg ? acc11 : acc01;
    v8h r0v, r1v;
    #pragma unroll
    for (int i = 0; i < 8; ++i) {
      float bi = bias[cb + i];
      r0v[i] = (_Float16)quantf(fmaxf(a0r[i] + bi, 0.0f));
      r1v[i] = (_Float16)quantf(fmaxf(a1r[i] + bi, 0.0f));
    }
    *(v8h*)(out + po0 + cb) = r0v;
    *(v8h*)(out + po1 + cb) = r1v;
  }
}

// ---------------------------------------------------------------------------
// FC via WMMA, 2x2 register tile: 2 Cout groups x 2 batch subtiles.
// A = repacked weights (M=cout), B = activations (N=batch).
// grid: (CoutP/32, B/32). mode 0: f16 bias+relu+quant ; mode 1: fp32, co<Cout.
// ---------------------------------------------------------------------------
__global__ __launch_bounds__(32) void fc_wmma(
    const _Float16* __restrict__ in, const _Float16* __restrict__ wpk,
    const float* __restrict__ bias, void* __restrict__ outv,
    int K, int Cout, int mode) {
  const int lane = threadIdx.x;
  const int g0 = blockIdx.x * 2;
  const int row0 = blockIdx.y * 32 + (lane & 15);
  const int row1 = row0 + 16;
  const int koff = (lane >> 4) << 4;
  const int kch = K >> 5;

  v8f acc00 = {}, acc01 = {}, acc10 = {}, acc11 = {};
  const long gstride = (long)kch * 512;
  const _Float16* aptr0 = wpk + (long)g0 * gstride + lane * 16;
  const _Float16* aptr1 = aptr0 + gstride;
  const _Float16* bptr0 = in + (long)row0 * K + koff;
  const _Float16* bptr1 = in + (long)row1 * K + koff;
  for (int cc = 0; cc < kch; ++cc) {
    v16h a0  = *(const v16h*)aptr0;  aptr0 += 512;
    v16h a1  = *(const v16h*)aptr1;  aptr1 += 512;
    v16h bf0 = *(const v16h*)(bptr0 + ((long)cc << 5));
    v16h bf1 = *(const v16h*)(bptr1 + ((long)cc << 5));
    acc00 = wmma16(a0, bf0, acc00);
    acc01 = wmma16(a0, bf1, acc01);
    acc10 = wmma16(a1, bf0, acc10);
    acc11 = wmma16(a1, bf1, acc11);
  }
  const int moff = (lane >> 4) << 3;
  #pragma unroll
  for (int gg = 0; gg < 2; ++gg) {
    const int cb = (g0 + gg) * 16 + moff;
    const v8f& a0r = gg ? acc10 : acc00;
    const v8f& a1r = gg ? acc11 : acc01;
    if (mode == 0) {
      _Float16* out = (_Float16*)outv;
      v8h r0v, r1v;
      #pragma unroll
      for (int i = 0; i < 8; ++i) {
        float bi = bias[cb + i];
        r0v[i] = (_Float16)quantf(fmaxf(a0r[i] + bi, 0.0f));
        r1v[i] = (_Float16)quantf(fmaxf(a1r[i] + bi, 0.0f));
      }
      *(v8h*)(out + (long)row0 * Cout + cb) = r0v;
      *(v8h*)(out + (long)row1 * Cout + cb) = r1v;
    } else {
      float* out = (float*)outv;
      #pragma unroll
      for (int i = 0; i < 8; ++i) {
        int co = cb + i;
        if (co < Cout) {
          float bi = bias[co];
          out[(long)row0 * Cout + co] = a0r[i] + bi;
          out[(long)row1 * Cout + co] = a1r[i] + bi;
        }
      }
    }
  }
}

// ---------------------------------------------------------------------------
// 2x2 max pool, padded NHWC in [B,2Ho+2,2Wo+2,C] -> padded out [B,Ho+2,Wo+2,C]
// ---------------------------------------------------------------------------
__global__ void maxpool2_nhwc(const _Float16* __restrict__ in, _Float16* __restrict__ out,
                              int Ho, int Wo, int C, long total) {
  long idx = (long)blockIdx.x * 256 + threadIdx.x;
  if (idx >= total) return;
  int c = (int)(idx % C); long t = idx / C;
  int xo = (int)(t % Wo); t /= Wo;
  int yo = (int)(t % Ho);
  int b  = (int)(t / Ho);
  int Hip = 2 * Ho + 2, Wip = 2 * Wo + 2;
  long base = (((long)b * Hip + 2 * yo + 1) * Wip + 2 * xo + 1) * C + c;
  float m0 = (float)in[base];
  float m1 = (float)in[base + C];
  float m2 = (float)in[base + (long)Wip * C];
  float m3 = (float)in[base + (long)Wip * C + C];
  out[(((long)b * (Ho + 2) + yo + 1) * (Wo + 2) + xo + 1) * C + c] =
      (_Float16)fmaxf(fmaxf(m0, m1), fmaxf(m2, m3));
}

// ---------------------------------------------------------------------------
// Adaptive avg pool 8x8 -> 3x3 (torch bin rule) + NCHW-order flatten.
// in: padded NHWC [256,10,10,256] f16 ; out: [256, 2304] f16, k = c*9 + i*3 + j
// ---------------------------------------------------------------------------
__global__ void adpool_flatten(const _Float16* __restrict__ in, _Float16* __restrict__ out) {
  int idx = blockIdx.x * 256 + threadIdx.x;   // 256*2304
  if (idx >= 256 * 2304) return;
  int k = idx % 2304; int b = idx / 2304;
  int c = k / 9; int ij = k % 9; int i = ij / 3; int j = ij % 3;
  const int s_[3] = {0, 2, 5};
  const int e_[3] = {3, 6, 8};
  float sum = 0.0f;
  for (int yy = s_[i]; yy < e_[i]; ++yy)
    for (int xx = s_[j]; xx < e_[j]; ++xx)
      sum += (float)in[(((long)b * 10 + yy + 1) * 10 + xx + 1) * 256 + c];
  float cnt = (float)((e_[i] - s_[i]) * (e_[j] - s_[j]));
  out[idx] = (_Float16)(sum / cnt);
}

// ---------------------------------------------------------------------------
extern "C" void kernel_launch(void* const* d_in, const int* in_sizes, int n_in,
                              void* d_out, int out_size, void* d_ws, size_t ws_size,
                              hipStream_t stream) {
  const float* x   = (const float*)d_in[0];
  const float* w1  = (const float*)d_in[1];
  const float* b1  = (const float*)d_in[2];
  const float* w2  = (const float*)d_in[3];
  const float* b2  = (const float*)d_in[4];
  const float* w3  = (const float*)d_in[5];
  const float* b3  = (const float*)d_in[6];
  const float* w4  = (const float*)d_in[7];
  const float* b4  = (const float*)d_in[8];
  const float* w5  = (const float*)d_in[9];
  const float* b5  = (const float*)d_in[10];
  const float* fw1 = (const float*)d_in[11];
  const float* fb1 = (const float*)d_in[12];
  const float* fw2 = (const float*)d_in[13];
  const float* fb2 = (const float*)d_in[14];
  const float* fw3 = (const float*)d_in[15];
  const float* fb3 = (const float*)d_in[16];

  char* ws = (char*)d_ws;
  // Region A (142.7 MB): a1p/a2p/a3p/a5p/flat/f2o
  // Region B ( 42.5 MB): p1p/p2p/a4p/p5p/f1o
  _Float16* regA = (_Float16*)(ws);
  _Float16* regB = (_Float16*)(ws + 142737408ull);          // 256*66*66*64*2
  size_t off = 142737408ull + 42467328ull;                  // + 256*18*18*256*2
  auto alloc = [&](size_t bytes) -> _Float16* {
    _Float16* p = (_Float16*)(ws + off);
    off += (bytes + 255) & ~(size_t)255;
    return p;
  };
  _Float16* wpk2 = alloc((size_t)192 * 64 * 9 * 2);
  _Float16* wpk3 = alloc((size_t)384 * 192 * 9 * 2);
  _Float16* wpk4 = alloc((size_t)256 * 384 * 9 * 2);
  _Float16* wpk5 = alloc((size_t)256 * 256 * 9 * 2);
  _Float16* fpk1 = alloc((size_t)4096 * 2304 * 2);
  _Float16* fpk2 = alloc((size_t)4096 * 4096 * 2);
  _Float16* fpk3 = alloc((size_t)32 * 4096 * 2);

  // --- weight repacks (deterministic, depend only on inputs) ---
  repack_conv_w<<<dim3(432),   dim3(256), 0, stream>>>(w2, wpk2, 64,  192);
  repack_conv_w<<<dim3(2592),  dim3(256), 0, stream>>>(w3, wpk3, 192, 384);
  repack_conv_w<<<dim3(3456),  dim3(256), 0, stream>>>(w4, wpk4, 384, 256);
  repack_conv_w<<<dim3(2304),  dim3(256), 0, stream>>>(w5, wpk5, 256, 256);
  repack_fc_w  <<<dim3(36864), dim3(256), 0, stream>>>(fw1, fpk1, 2304, 4096, 4096);
  repack_fc_w  <<<dim3(65536), dim3(256), 0, stream>>>(fw2, fpk2, 4096, 4096, 4096);
  repack_fc_w  <<<dim3(512),   dim3(256), 0, stream>>>(fw3, fpk3, 4096, 10,   32);

  // --- conv1 -> A : padded [256,66,66,64] ---
  zero_border<<<dim3(16640), dim3(256), 0, stream>>>(regA, 64, 64, 64, 4259840L);
  conv1_direct<<<dim3(262144), dim3(256), 0, stream>>>(x, w1, b1, regA);
  // pool1 -> B : padded [256,34,34,64]
  zero_border<<<dim3(8448), dim3(256), 0, stream>>>(regB, 32, 32, 64, 2162688L);
  maxpool2_nhwc<<<dim3(65536), dim3(256), 0, stream>>>(regA, regB, 32, 32, 64, 16777216L);
  // conv2 -> A : padded [256,34,34,192]
  zero_border<<<dim3(25344), dim3(256), 0, stream>>>(regA, 32, 32, 192, 6488064L);
  conv3x3_wmma<<<dim3(8192, 6), dim3(32), 0, stream>>>(regB, wpk2, b2, regA, 32, 32, 64, 192);
  // pool2 -> B : padded [256,18,18,192]
  zero_border<<<dim3(13056), dim3(256), 0, stream>>>(regB, 16, 16, 192, 3342336L);
  maxpool2_nhwc<<<dim3(49152), dim3(256), 0, stream>>>(regA, regB, 16, 16, 192, 12582912L);
  // conv3 -> A : padded [256,18,18,384]
  zero_border<<<dim3(26112), dim3(256), 0, stream>>>(regA, 16, 16, 384, 6684672L);
  conv3x3_wmma<<<dim3(2048, 12), dim3(32), 0, stream>>>(regB, wpk3, b3, regA, 16, 16, 192, 384);
  // conv4 -> B : padded [256,18,18,256]
  zero_border<<<dim3(17408), dim3(256), 0, stream>>>(regB, 16, 16, 256, 4456448L);
  conv3x3_wmma<<<dim3(2048, 8), dim3(32), 0, stream>>>(regA, wpk4, b4, regB, 16, 16, 384, 256);
  // conv5 -> A : padded [256,18,18,256]
  zero_border<<<dim3(17408), dim3(256), 0, stream>>>(regA, 16, 16, 256, 4456448L);
  conv3x3_wmma<<<dim3(2048, 8), dim3(32), 0, stream>>>(regB, wpk5, b5, regA, 16, 16, 256, 256);
  // pool5 -> B : padded [256,10,10,256]
  zero_border<<<dim3(9216), dim3(256), 0, stream>>>(regB, 8, 8, 256, 2359296L);
  maxpool2_nhwc<<<dim3(16384), dim3(256), 0, stream>>>(regA, regB, 8, 8, 256, 4194304L);
  // adaptive pool + flatten -> A : [256,2304]
  adpool_flatten<<<dim3(2304), dim3(256), 0, stream>>>(regB, regA);
  // fc1 -> B : [256,4096]
  fc_wmma<<<dim3(128, 8), dim3(32), 0, stream>>>(regA, fpk1, fb1, (void*)regB, 2304, 4096, 0);
  // fc2 -> A : [256,4096]
  fc_wmma<<<dim3(128, 8), dim3(32), 0, stream>>>(regB, fpk2, fb2, (void*)regA, 4096, 4096, 0);
  // fc3 -> d_out : [256,10] fp32
  fc_wmma<<<dim3(1, 8), dim3(32), 0, stream>>>(regA, fpk3, fb3, d_out, 4096, 10, 1);
}
